// Block_64862596104350
// MI455X (gfx1250) — compile-verified, matrix operand order checked by reading, not run on hardware
//
#include <hip/hip_runtime.h>

typedef __attribute__((ext_vector_type(16))) __bf16 v16bf;
typedef __attribute__((ext_vector_type(8)))  float  v8f;
typedef __attribute__((ext_vector_type(4)))  int    v4i;

#define DIMX 2048
#define NH   16
#define DQKX 192
#define DNOPEX 128
#define DVX  128
#define NEXP 16
#define MHID 1024
#define EPSX 1e-5f

#if defined(__HIP_DEVICE_COMPILE__) && __has_builtin(__builtin_amdgcn_global_load_async_to_lds_b128) && __has_builtin(__builtin_amdgcn_s_wait_asynccnt)
#define HAVE_ASYNC_LDS 1
#else
#define HAVE_ASYNC_LDS 0
#endif
#define AS1 __attribute__((address_space(1)))
#define AS3 __attribute__((address_space(3)))

static __device__ __forceinline__ unsigned short f2bf(float f) {
  unsigned u = __float_as_uint(f);
  unsigned r = (u + 0x7FFFu + ((u >> 16) & 1u)) >> 16;
  return (unsigned short)r;
}
static __device__ __forceinline__ unsigned pack_bf(float a, float b) {
  return (unsigned)f2bf(a) | ((unsigned)f2bf(b) << 16);
}
static __device__ __forceinline__ float bf2f(unsigned short h) {
  return __uint_as_float(((unsigned)h) << 16);
}

// ---------------- RMSNorm -> bf16 ----------------
__global__ __launch_bounds__(256) void rmsnorm_bf16_kernel(
    const float* __restrict__ x, const float* __restrict__ w,
    unsigned short* __restrict__ out) {
  int row = blockIdx.x;
  const float* xr = x + (size_t)row * DIMX;
  __shared__ float red[256];
  float s = 0.f;
  for (int i = threadIdx.x; i < DIMX; i += 256) { float v = xr[i]; s += v * v; }
  red[threadIdx.x] = s; __syncthreads();
  for (int off = 128; off > 0; off >>= 1) {
    if ((int)threadIdx.x < off) red[threadIdx.x] += red[threadIdx.x + off];
    __syncthreads();
  }
  float scale = rsqrtf(red[0] / (float)DIMX + EPSX);
  unsigned short* o = out + (size_t)row * DIMX;
  for (int i = threadIdx.x; i < DIMX; i += 256) o[i] = f2bf(xr[i] * scale * w[i]);
}

// ---------------- WMMA bf16 GEMM: C[M,N] = A[M,K](bf16) @ W[N,K]^T(f32->bf16) ----------------
// 64x128 macro-tile, BK=32, 8 waves in 2(M)x4(N), 2x2 16x16 sub-tiles per wave.
// A tile streamed via GLOBAL_LOAD_ASYNC_TO_LDS_B128 (ASYNCcnt); W converted in VGPRs.
__global__ __launch_bounds__(256) void gemm_bf16_kernel(
    const unsigned short* __restrict__ A, const float* __restrict__ W,
    float* __restrict__ C, int M, int N, int K,
    const float* __restrict__ bias,
    const float* __restrict__ row_scale, int rs_stride,
    const float* __restrict__ residual, int accumulate) {
  __shared__ __align__(16) unsigned short lds_a[64 * 40];
  __shared__ __align__(16) unsigned short lds_b[128 * 40];
  const int tid  = threadIdx.x;
  const int lane = tid & 31;
  const int wid  = tid >> 5;
  const int wm = wid >> 2, wn = wid & 3;          // 2 x 4 wave grid
  const int m0 = blockIdx.y * 64;
  const int n0 = blockIdx.x * 128;
  v8f acc[2][2];
  const v8f vzero = {};
#pragma unroll
  for (int mi = 0; mi < 2; ++mi)
#pragma unroll
    for (int ni = 0; ni < 2; ++ni) acc[mi][ni] = vzero;
  const int nk = K >> 5;
  const int ar = tid >> 2, ac = (tid & 3) * 8;    // A: 64x32 bf16, 16B/thread
  const int wr = tid >> 1, wc = (tid & 1) * 16;   // W: 128x32 f32, 64B/thread
  const int hi = (lane >> 4) << 3;
  for (int kk = 0; kk < nk; ++kk) {
    const int k0 = kk << 5;
#if HAVE_ASYNC_LDS
    __builtin_amdgcn_global_load_async_to_lds_b128(
        (AS1 v4i*)(A + (size_t)(m0 + ar) * K + k0 + ac),
        (AS3 v4i*)&lds_a[ar * 40 + ac], 0, 0);
#else
    {
      const uint4 av = *reinterpret_cast<const uint4*>(A + (size_t)(m0 + ar) * K + k0 + ac);
      *reinterpret_cast<uint4*>(&lds_a[ar * 40 + ac]) = av;
    }
#endif
    { // W tile (f32 -> bf16): 128x32, 16 floats/thread
      const float* wp = W + (size_t)(n0 + wr) * K + k0 + wc;
      const float4 w0 = *reinterpret_cast<const float4*>(wp);
      const float4 w1 = *reinterpret_cast<const float4*>(wp + 4);
      const float4 w2 = *reinterpret_cast<const float4*>(wp + 8);
      const float4 w3 = *reinterpret_cast<const float4*>(wp + 12);
      uint4 p0, p1;
      p0.x = pack_bf(w0.x, w0.y); p0.y = pack_bf(w0.z, w0.w);
      p0.z = pack_bf(w1.x, w1.y); p0.w = pack_bf(w1.z, w1.w);
      p1.x = pack_bf(w2.x, w2.y); p1.y = pack_bf(w2.z, w2.w);
      p1.z = pack_bf(w3.x, w3.y); p1.w = pack_bf(w3.z, w3.w);
      *reinterpret_cast<uint4*>(&lds_b[wr * 40 + wc]) = p0;
      *reinterpret_cast<uint4*>(&lds_b[wr * 40 + wc + 8]) = p1;
      if (kk + 1 < nk) __builtin_prefetch(wp + 32, 0, 1);  // global_prefetch_b8 next K-tile
    }
#if HAVE_ASYNC_LDS
    __builtin_amdgcn_s_wait_asynccnt(0);
#endif
    __syncthreads();
    union F { int i[8]; v16bf v; };
    F au[2], bu[2];
#pragma unroll
    for (int mi = 0; mi < 2; ++mi) {
      const int am = wm * 32 + mi * 16 + (lane & 15);
#pragma unroll
      for (int vv = 0; vv < 8; ++vv) {
        const int kb = hi + ((vv & 3) << 1) + ((vv >> 2) << 4);
        au[mi].i[vv] = *reinterpret_cast<const unsigned*>(&lds_a[am * 40 + kb]);
      }
    }
#pragma unroll
    for (int ni = 0; ni < 2; ++ni) {
      const int bn = wn * 32 + ni * 16 + (lane & 15);
#pragma unroll
      for (int vv = 0; vv < 8; ++vv) {
        const int kb = hi + ((vv & 3) << 1) + ((vv >> 2) << 4);
        bu[ni].i[vv] = *reinterpret_cast<const unsigned*>(&lds_b[bn * 40 + kb]);
      }
    }
#pragma unroll
    for (int mi = 0; mi < 2; ++mi)
#pragma unroll
      for (int ni = 0; ni < 2; ++ni)
        acc[mi][ni] = __builtin_amdgcn_wmma_f32_16x16x32_bf16(
            false, au[mi].v, false, bu[ni].v, (short)0, acc[mi][ni], false, false);
    __syncthreads();
  }
#pragma unroll
  for (int mi = 0; mi < 2; ++mi) {
    const int mb = m0 + wm * 32 + mi * 16 + hi;
#pragma unroll
    for (int ni = 0; ni < 2; ++ni) {
      const int gn = n0 + wn * 32 + ni * 16 + (lane & 15);
#pragma unroll
      for (int vv = 0; vv < 8; ++vv) {
        const int gm = mb + vv;
        float val = acc[mi][ni][vv];
        if (row_scale) val *= row_scale[(size_t)gm * rs_stride];
        if (bias) val += bias[gn];
        const size_t off = (size_t)gm * N + gn;
        if (residual) val += residual[off];
        if (accumulate) val += C[off];
        C[off] = val;
      }
    }
  }
}

// ---------------- RoPE (in place on q and k) ----------------
__global__ void rope_kernel(float* __restrict__ q, float* __restrict__ k,
                            const int* __restrict__ pos, int n) {
  int idx = blockIdx.x * blockDim.x + threadIdx.x;
  if (idx >= n) return;
  int i  = idx & 31;
  int hh = (idx >> 5) & 15;
  int t  = idx >> 9;
  float p  = (float)pos[t];
  float th = __powf(10000.f, -(float)(2 * i) / 64.f);
  float sv, cvv; __sincosf(p * th, &sv, &cvv);
  size_t base = (size_t)t * (NH * DQKX) + hh * DQKX + DNOPEX + 2 * i;
  float q0 = q[base], q1 = q[base + 1];
  q[base] = q0 * cvv - q1 * sv; q[base + 1] = q1 * cvv + q0 * sv;
  float k0 = k[base], k1 = k[base + 1];
  k[base] = k0 * cvv - k1 * sv; k[base + 1] = k1 * cvv + k0 * sv;
}

// ---------------- Flash attention: 1 wave per (head, 16-query tile) ----------------
__global__ __launch_bounds__(32) void flash_attn_kernel(
    const float* __restrict__ q, const float* __restrict__ k, const float* __restrict__ v,
    unsigned short* __restrict__ cv, int S) {
  const int lane = threadIdx.x;
  const int q0   = blockIdx.x * 16;
  const int head = blockIdx.y;
  const float scale = 0.07216878365f;  // 1/sqrt(192)
  __shared__ __align__(16) unsigned short lds_p[16 * 36];
  union F { int i[8]; v16bf v; };
  const int hi = (lane >> 4) << 3;
  const int m  = q0 + (lane & 15);
  F qf[6];
#pragma unroll
  for (int ch = 0; ch < 6; ++ch)
#pragma unroll
    for (int vv = 0; vv < 8; ++vv) {
      const int c = ch * 32 + hi + ((vv & 3) << 1) + ((vv >> 2) << 4);
      const float2 f = *reinterpret_cast<const float2*>(q + (size_t)m * (NH * DQKX) + head * DQKX + c);
      qf[ch].i[vv] = pack_bf(f.x, f.y);
    }
  v8f o[8]; const v8f vzero = {};
#pragma unroll
  for (int n = 0; n < 8; ++n) o[n] = vzero;
  float mst[8], lst[8];
#pragma unroll
  for (int vv = 0; vv < 8; ++vv) { mst[vv] = -1e30f; lst[vv] = 0.f; }

  for (int j0 = 0; j0 < S; j0 += 32) {
    v8f s0 = vzero, s1 = vzero;
    const int key0 = j0 + (lane & 15);
    const int key1 = key0 + 16;
#pragma unroll
    for (int ch = 0; ch < 6; ++ch) {
      F kb0, kb1;
#pragma unroll
      for (int vv = 0; vv < 8; ++vv) {
        const int c = ch * 32 + hi + ((vv & 3) << 1) + ((vv >> 2) << 4);
        const float2 f0 = *reinterpret_cast<const float2*>(k + (size_t)key0 * (NH * DQKX) + head * DQKX + c);
        const float2 f1 = *reinterpret_cast<const float2*>(k + (size_t)key1 * (NH * DQKX) + head * DQKX + c);
        kb0.i[vv] = pack_bf(f0.x, f0.y);
        kb1.i[vv] = pack_bf(f1.x, f1.y);
      }
      s0 = __builtin_amdgcn_wmma_f32_16x16x32_bf16(false, qf[ch].v, false, kb0.v, (short)0, s0, false, false);
      s1 = __builtin_amdgcn_wmma_f32_16x16x32_bf16(false, qf[ch].v, false, kb1.v, (short)0, s1, false, false);
    }
#pragma unroll
    for (int vv = 0; vv < 8; ++vv) {
      float a = s0[vv] * scale, b = s1[vv] * scale;
      float rm = fmaxf(a, b);
      rm = fmaxf(rm, __shfl_xor(rm, 1, 32));
      rm = fmaxf(rm, __shfl_xor(rm, 2, 32));
      rm = fmaxf(rm, __shfl_xor(rm, 4, 32));
      rm = fmaxf(rm, __shfl_xor(rm, 8, 32));
      const float mnew  = fmaxf(mst[vv], rm);
      const float alpha = __expf(mst[vv] - mnew);
      const float p0 = __expf(a - mnew);
      const float p1 = __expf(b - mnew);
      float rs = p0 + p1;
      rs += __shfl_xor(rs, 1, 32);
      rs += __shfl_xor(rs, 2, 32);
      rs += __shfl_xor(rs, 4, 32);
      rs += __shfl_xor(rs, 8, 32);
      lst[vv] = lst[vv] * alpha + rs;
      mst[vv] = mnew;
#pragma unroll
      for (int n = 0; n < 8; ++n) o[n][vv] *= alpha;
      const int row = vv + hi;
      lds_p[row * 36 + (lane & 15)]      = f2bf(p0);
      lds_p[row * 36 + 16 + (lane & 15)] = f2bf(p1);
    }
    __syncthreads();
    F pf;
#pragma unroll
    for (int vv = 0; vv < 8; ++vv) {
      const int kb = hi + ((vv & 3) << 1) + ((vv >> 2) << 4);
      pf.i[vv] = *reinterpret_cast<const unsigned*>(&lds_p[(lane & 15) * 36 + kb]);
    }
#pragma unroll
    for (int n = 0; n < 8; ++n) {
      F vf;
      const int dv = n * 16 + (lane & 15);
#pragma unroll
      for (int vv = 0; vv < 8; ++vv) {
        const int kb = hi + ((vv & 3) << 1) + ((vv >> 2) << 4);
        const float a = v[(size_t)(j0 + kb)     * (NH * DVX) + head * DVX + dv];
        const float b = v[(size_t)(j0 + kb + 1) * (NH * DVX) + head * DVX + dv];
        vf.i[vv] = pack_bf(a, b);
      }
      o[n] = __builtin_amdgcn_wmma_f32_16x16x32_bf16(false, pf.v, false, vf.v, (short)0, o[n], false, false);
    }
    __syncthreads();
  }
  const int mb = q0 + hi;
#pragma unroll
  for (int vv = 0; vv < 8; ++vv) {
    const float inv = 1.f / lst[vv];
    const int t = mb + vv;
#pragma unroll
    for (int n = 0; n < 8; ++n) {
      const int dv = n * 16 + (lane & 15);
      cv[(size_t)t * (NH * DVX) + head * DVX + dv] = f2bf(o[n][vv] * inv);
    }
  }
}

// ---------------- Gate: softmax over 16 logits, top-2 -> dense cw[T,16] ----------------
__global__ __launch_bounds__(256) void gate_topk_kernel(
    const unsigned short* __restrict__ h, const float* __restrict__ gw,
    float* __restrict__ cw) {
  const int t = blockIdx.x;
  __shared__ float part[256];
  __shared__ float logits[16];
  const int e = threadIdx.x & 15;
  const int chunk = threadIdx.x >> 4;
  float acc = 0.f;
  const unsigned short* hr = h + (size_t)t * DIMX + chunk * 128;
  const float* wr = gw + (size_t)e * DIMX + chunk * 128;
  for (int i = 0; i < 128; ++i) acc += bf2f(hr[i]) * wr[i];
  part[threadIdx.x] = acc;
  __syncthreads();
  if (threadIdx.x < 16) {
    float s = 0.f;
    for (int c = 0; c < 16; ++c) s += part[c * 16 + threadIdx.x];
    logits[threadIdx.x] = s;
  }
  __syncthreads();
  if (threadIdx.x == 0) {
    float mx = -1e30f;
    for (int i = 0; i < 16; ++i) mx = fmaxf(mx, logits[i]);
    float se = 0.f, p[16];
    for (int i = 0; i < 16; ++i) { p[i] = __expf(logits[i] - mx); se += p[i]; }
    const float inv = 1.f / se;
    int i1 = 0; float b1 = -1.f;
    for (int i = 0; i < 16; ++i) { p[i] *= inv; if (p[i] > b1) { b1 = p[i]; i1 = i; } }
    int i2 = 0; float b2 = -1.f;
    for (int i = 0; i < 16; ++i) if (i != i1 && p[i] > b2) { b2 = p[i]; i2 = i; }
    for (int i = 0; i < 16; ++i) cw[(size_t)t * 16 + i] = 0.f;
    cw[(size_t)t * 16 + i1] = b1;  // ROUTE_SCALE = 1.0
    cw[(size_t)t * 16 + i2] = b2;
  }
}

// ---------------- SiLU(g) * u -> bf16 ----------------
__global__ void silu_mul_kernel(const float* __restrict__ g, const float* __restrict__ u,
                                unsigned short* __restrict__ out, int n) {
  int i = blockIdx.x * blockDim.x + threadIdx.x;
  if (i >= n) return;
  float x = g[i];
  out[i] = f2bf((x / (1.f + __expf(-x))) * u[i]);
}

__global__ void add2_kernel(const float* __restrict__ a, const float* __restrict__ b,
                            float* __restrict__ o, int n) {
  int i = blockIdx.x * blockDim.x + threadIdx.x;
  if (i < n) o[i] = a[i] + b[i];
}

extern "C" void kernel_launch(void* const* d_in, const int* in_sizes, int n_in,
                              void* d_out, int out_size, void* d_ws, size_t ws_size,
                              hipStream_t stream) {
  const float* x      = (const float*)d_in[0];
  const int*   pos    = (const int*)  d_in[1];
  const float* anw    = (const float*)d_in[3];
  const float* fnw    = (const float*)d_in[4];
  const float* wq     = (const float*)d_in[5];
  const float* wk     = (const float*)d_in[6];
  const float* wv     = (const float*)d_in[7];
  const float* cv_w   = (const float*)d_in[8];
  const float* cv_b   = (const float*)d_in[9];
  const float* gate_w = (const float*)d_in[10];
  const float* e_w1   = (const float*)d_in[11];
  const float* e_w2   = (const float*)d_in[12];
  const float* e_w3   = (const float*)d_in[13];
  const float* s_w1   = (const float*)d_in[14];
  const float* s_w2   = (const float*)d_in[15];
  const float* s_w3   = (const float*)d_in[16];
  const int T = in_sizes[0] / DIMX;   // 2048 tokens

  char* p = (char*)d_ws;
  auto alloc = [&](size_t bytes) { char* r = p; p += (bytes + 255) & ~(size_t)255; return r; };
  unsigned short* h1b = (unsigned short*)alloc((size_t)T * DIMX * 2);
  float* qb  = (float*)alloc((size_t)T * NH * DQKX * 4);
  float* kb  = (float*)alloc((size_t)T * NH * DQKX * 4);
  float* vb  = (float*)alloc((size_t)T * NH * DVX * 4);
  unsigned short* cvb = (unsigned short*)alloc((size_t)T * NH * DVX * 2);
  float* x1  = (float*)alloc((size_t)T * DIMX * 4);
  unsigned short* h2b = (unsigned short*)alloc((size_t)T * DIMX * 2);
  float* cw  = (float*)alloc((size_t)T * NEXP * 4);
  float* t1  = (float*)alloc((size_t)T * MHID * 4);
  float* t2  = (float*)alloc((size_t)T * MHID * 4);
  unsigned short* a2 = (unsigned short*)alloc((size_t)T * MHID * 2);
  float* yb  = (float*)alloc((size_t)T * DIMX * 4);

  const dim3 blk(256);
  // attention-input RMSNorm -> bf16
  rmsnorm_bf16_kernel<<<T, 256, 0, stream>>>(x, anw, h1b);
  // q, k, v projections
  gemm_bf16_kernel<<<dim3((NH * DQKX) / 128, T / 64), blk, 0, stream>>>(h1b, wq, qb, T, NH * DQKX, DIMX, nullptr, nullptr, 0, nullptr, 0);
  gemm_bf16_kernel<<<dim3((NH * DQKX) / 128, T / 64), blk, 0, stream>>>(h1b, wk, kb, T, NH * DQKX, DIMX, nullptr, nullptr, 0, nullptr, 0);
  gemm_bf16_kernel<<<dim3((NH * DVX) / 128, T / 64), blk, 0, stream>>>(h1b, wv, vb, T, NH * DVX, DIMX, nullptr, nullptr, 0, nullptr, 0);
  // RoPE
  {
    int n = T * NH * 32;
    rope_kernel<<<(n + 255) / 256, 256, 0, stream>>>(qb, kb, pos, n);
  }
  // flash attention -> cv (bf16)
  flash_attn_kernel<<<dim3(T / 16, NH), 32, 0, stream>>>(qb, kb, vb, cvb, T);
  // output projection + bias + residual with x
  gemm_bf16_kernel<<<dim3(DIMX / 128, T / 64), blk, 0, stream>>>(cvb, cv_w, x1, T, DIMX, NH * DVX, cv_b, nullptr, 0, x, 0);
  // FFN RMSNorm -> bf16
  rmsnorm_bf16_kernel<<<T, 256, 0, stream>>>(x1, fnw, h2b);
  // router
  gate_topk_kernel<<<T, 256, 0, stream>>>(h2b, gate_w, cw);
  // shared expert -> y (fresh write)
  gemm_bf16_kernel<<<dim3(MHID / 128, T / 64), blk, 0, stream>>>(h2b, s_w1, t1, T, MHID, DIMX, nullptr, nullptr, 0, nullptr, 0);
  gemm_bf16_kernel<<<dim3(MHID / 128, T / 64), blk, 0, stream>>>(h2b, s_w3, t2, T, MHID, DIMX, nullptr, nullptr, 0, nullptr, 0);
  {
    int n = T * MHID;
    silu_mul_kernel<<<(n + 255) / 256, 256, 0, stream>>>(t1, t2, a2, n);
  }
  gemm_bf16_kernel<<<dim3(DIMX / 128, T / 64), blk, 0, stream>>>(a2, s_w2, yb, T, DIMX, MHID, nullptr, nullptr, 0, nullptr, 0);
  // routed experts: accumulate cw[:,e]-scaled outputs into y
  for (int e = 0; e < NEXP; ++e) {
    const float* w1 = e_w1 + (size_t)e * MHID * DIMX;
    const float* w3 = e_w3 + (size_t)e * MHID * DIMX;
    const float* w2 = e_w2 + (size_t)e * DIMX * MHID;
    gemm_bf16_kernel<<<dim3(MHID / 128, T / 64), blk, 0, stream>>>(h2b, w1, t1, T, MHID, DIMX, nullptr, nullptr, 0, nullptr, 0);
    gemm_bf16_kernel<<<dim3(MHID / 128, T / 64), blk, 0, stream>>>(h2b, w3, t2, T, MHID, DIMX, nullptr, nullptr, 0, nullptr, 0);
    int n = T * MHID;
    silu_mul_kernel<<<(n + 255) / 256, 256, 0, stream>>>(t1, t2, a2, n);
    gemm_bf16_kernel<<<dim3(DIMX / 128, T / 64), blk, 0, stream>>>(a2, w2, yb, T, DIMX, MHID, nullptr, cw + e, NEXP, nullptr, 1);
  }
  // out = x1 + y
  {
    int n = T * DIMX;
    add2_kernel<<<(n + 255) / 256, 256, 0, stream>>>(x1, yb, (float*)d_out, n);
  }
}
